// GraniteSpeechConformerAttention_14319420965045
// MI455X (gfx1250) — compile-verified, matrix operand order checked by reading, not run on hardware
//
#include <hip/hip_runtime.h>

// ---------------------------------------------------------------------------
// Types / helpers
// ---------------------------------------------------------------------------
typedef __attribute__((ext_vector_type(16))) __bf16 v16bf;
typedef __attribute__((ext_vector_type(8)))  float  v8f;
typedef __attribute__((ext_vector_type(4)))  unsigned int u32x4;
typedef __attribute__((ext_vector_type(8)))  int i32x8;
typedef __attribute__((ext_vector_type(4)))  int i32x4;

struct alignas(16) U128 { unsigned int x[4]; };
struct alignas(8)  U64  { unsigned int x[2]; };

union FragU { U128 q[2]; v16bf v; };

#if defined(__has_builtin)
#if __has_builtin(__builtin_amdgcn_tensor_load_to_lds) && \
    __has_builtin(__builtin_amdgcn_s_wait_tensorcnt)
#define HAVE_TDM 1
#endif
#endif
#ifndef HAVE_TDM
#define HAVE_TDM 0
#endif

// Load a 16x32 bf16 A-fragment (or mirrored B-fragment) per the CDNA5 layout:
// lane half h = lane>>4, element e -> K = 16*(e>>3) + 8*h + (e&7)
// p0 must already include the row offset (row-major, contiguous in K).
static __device__ inline v16bf load_frag(const __bf16* p0, int k0, int lane) {
  int h = (lane >> 4) & 1;
  const __bf16* p = p0 + k0 + 8 * h;
  FragU u;
  u.q[0] = *(const U128*)(p);
  u.q[1] = *(const U128*)(p + 16);
  return u.v;
}

static __device__ inline v8f wmma_bf16(v16bf a, v16bf b, v8f c) {
  return __builtin_amdgcn_wmma_f32_16x16x32_bf16(false, a, false, b, (short)0, c,
                                                 false, false);
}

#define N_TOK   32768     // B*S
#define DMODEL  1024
#define NHEAD   8
#define DHEAD   128
#define CSIZE   256
#define NBLK    16
#define NGRP    1024      // B*NB*H
#define SCALE_F 0.08838834764831845f

// ---------------------------------------------------------------------------
// Weight / table conversion to bf16 (transposed for contiguous B-fragments)
// ---------------------------------------------------------------------------
__global__ void cvt_wqkv(const float* __restrict__ Wq, const float* __restrict__ Wkv,
                         __bf16* __restrict__ o) {
  int idx = blockIdx.x * 256 + threadIdx.x;
  if (idx >= 3072 * 1024) return;
  int n = idx >> 10, k = idx & 1023;
  float v;
  if (n < 1024)       v = Wq[k * 1024 + n];
  else if (n < 2048)  v = Wkv[k * 2048 + (n - 1024)];
  else                v = Wkv[k * 2048 + 1024 + (n - 2048)];
  o[idx] = (__bf16)v;   // layout [n][k]
}

__global__ void cvt_wout(const float* __restrict__ W, __bf16* __restrict__ o) {
  int idx = blockIdx.x * 256 + threadIdx.x;
  if (idx >= 1024 * 1024) return;
  int n = idx >> 10, k = idx & 1023;
  o[idx] = (__bf16)W[k * 1024 + n];  // layout [n][k]
}

__global__ void cvt_rel(const float* __restrict__ r, __bf16* __restrict__ o, int n) {
  int idx = blockIdx.x * 256 + threadIdx.x;
  if (idx < n) o[idx] = (__bf16)r[idx];   // keep [row=dist][d] layout
}

// ---------------------------------------------------------------------------
// LayerNorm (one block = one row of 1024)
// ---------------------------------------------------------------------------
__global__ __launch_bounds__(256) void ln_kernel(const float* __restrict__ x,
                                                 const float* __restrict__ gamma,
                                                 const float* __restrict__ beta,
                                                 __bf16* __restrict__ out) {
  int row = blockIdx.x, t = threadIdx.x;
  const float4 xv = *(const float4*)(x + (size_t)row * DMODEL + t * 4);
  float xa[4] = {xv.x, xv.y, xv.z, xv.w};
  float s = xa[0] + xa[1] + xa[2] + xa[3];
  float s2 = xa[0] * xa[0] + xa[1] * xa[1] + xa[2] * xa[2] + xa[3] * xa[3];
#pragma unroll
  for (int m = 16; m >= 1; m >>= 1) {
    s  += __shfl_xor(s,  m, 32);
    s2 += __shfl_xor(s2, m, 32);
  }
  __shared__ float rs[8], rs2[8];
  int w = t >> 5, lane = t & 31;
  if (lane == 0) { rs[w] = s; rs2[w] = s2; }
  __syncthreads();
  if (t == 0) {
    float a = 0.f, b = 0.f;
#pragma unroll
    for (int i = 0; i < 8; i++) { a += rs[i]; b += rs2[i]; }
    rs[0] = a; rs2[0] = b;
  }
  __syncthreads();
  float mean = rs[0] * (1.0f / DMODEL);
  float var  = rs2[0] * (1.0f / DMODEL) - mean * mean;
  float inv  = rsqrtf(var + 1e-5f);
  __bf16 o[4];
#pragma unroll
  for (int i = 0; i < 4; i++) {
    float xf = (xa[i] - mean) * inv * gamma[t * 4 + i] + beta[t * 4 + i];
    o[i] = (__bf16)xf;
  }
  *(U64*)(out + (size_t)row * DMODEL + t * 4) = *(U64*)o;
}

// ---------------------------------------------------------------------------
// GEMM: xn[32768,1024] @ Wcat -> q/k/v (bf16, attention-friendly layouts)
// grid (256, 24), 256 threads (8 waves, 16 rows each), tile 128x128.
// Double-buffered: loads for k+32 are in flight during k's WMMAs.
// ---------------------------------------------------------------------------
__global__ __launch_bounds__(256) void gemm_qkv(const __bf16* __restrict__ xn,
                                                const __bf16* __restrict__ w,
                                                __bf16* __restrict__ qw,
                                                __bf16* __restrict__ kw,
                                                __bf16* __restrict__ vw) {
  int tid = threadIdx.x, lane = tid & 31, wv = tid >> 5;
  int m0 = blockIdx.x * 128 + wv * 16;
  int n0 = blockIdx.y * 128;
  int am = lane & 15, hl = lane >> 4;

  v8f acc[8];
#pragma unroll
  for (int nt = 0; nt < 8; nt++) acc[nt] = (v8f){0.f,0.f,0.f,0.f,0.f,0.f,0.f,0.f};

  const __bf16* arow = xn + (size_t)(m0 + am) * 1024;
  const __bf16* brow0 = w + (size_t)(n0 + am) * 1024;

  v16bf a_cur = load_frag(arow, 0, lane);
  v16bf b_cur[8];
#pragma unroll
  for (int nt = 0; nt < 8; nt++) b_cur[nt] = load_frag(brow0 + nt * 16 * 1024, 0, lane);

  for (int k0 = 0; k0 < 1024; k0 += 32) {
    // prefetch next k-step (tail over-read stays inside workspace)
    v16bf a_nxt = load_frag(arow, k0 + 32, lane);
    v16bf b_nxt[8];
#pragma unroll
    for (int nt = 0; nt < 8; nt++)
      b_nxt[nt] = load_frag(brow0 + nt * 16 * 1024, k0 + 32, lane);
#pragma unroll
    for (int nt = 0; nt < 8; nt++) acc[nt] = wmma_bf16(a_cur, b_cur[nt], acc[nt]);
    a_cur = a_nxt;
#pragma unroll
    for (int nt = 0; nt < 8; nt++) b_cur[nt] = b_nxt[nt];
  }
#pragma unroll
  for (int nt = 0; nt < 8; nt++) {
    int n = n0 + nt * 16 + am;
#pragma unroll
    for (int r = 0; r < 8; r++) {
      int m = m0 + r + 8 * hl;
      __bf16 bv = (__bf16)acc[nt][r];
      int b = m >> 12, srem = m & 4095;
      int blk = srem >> 8, i = srem & 255;
      if (n < 1024) {
        int hh = n >> 7, d = n & 127;
        qw[(size_t)(((b * 16 + blk) * 8 + hh) * 256 + i) * 128 + d] = bv;
      } else if (n < 2048) {
        int nn = n - 1024; int hh = nn >> 7, d = nn & 127;
        kw[(size_t)(((b * 16 + blk) * 8 + hh) * 256 + i) * 128 + d] = bv;
      } else {
        int nn = n - 2048; int hh = nn >> 7, d = nn & 127;
        vw[(size_t)((b * 16 + blk) * 8 + hh) * 32768 + d * 256 + i] = bv;  // V^T
      }
    }
  }
}

// ---------------------------------------------------------------------------
// Tensor Data Mover staging: one D# describing a contiguous 64KB block
// (16384 dwords) as a 1-D tile, loaded Global -> LDS.
// ---------------------------------------------------------------------------
#if HAVE_TDM
static __device__ inline void tdm_load_64kb(unsigned lds_off, const void* gptr) {
  unsigned long long ga = (unsigned long long)gptr;
  u32x4 g0;
  g0[0] = 1u;                                   // count=1 valid descriptor
  g0[1] = lds_off;                              // lds_addr (bytes)
  g0[2] = (unsigned)(ga & 0xFFFFFFFFu);         // global_addr[31:0]
  g0[3] = (unsigned)((ga >> 32) & 0x1FFFFFFu)   // global_addr[56:32]
          | (2u << 30);                         // type = 2 ("image")
  i32x8 g1;
  g1[0] = (int)(2u << 16);                      // data_size=2 (4B), mask=0
  g1[1] = (int)(16384u << 16);                  // tensor_dim0[15:0]<<16
  g1[2] = (int)(1u << 16);                      // dim0 hi=0 | tensor_dim1=1
  g1[3] = (int)(16384u << 16);                  // dim1 hi=0 | tile_dim0=16384
  g1[4] = 1;                                    // tile_dim1=1, tile_dim2=0
  g1[5] = 16384;                                // tensor_dim0_stride lo
  g1[6] = 0;                                    // stride0 hi | stride1 lo
  g1[7] = 0;                                    // stride1 hi
  i32x4 gz = (i32x4){0, 0, 0, 0};
#if __clang_major__ >= 23
  i32x8 gz8 = (i32x8){0, 0, 0, 0, 0, 0, 0, 0};
  __builtin_amdgcn_tensor_load_to_lds(g0, g1, gz, gz, gz8, 0);
#else
  __builtin_amdgcn_tensor_load_to_lds(g0, g1, gz, gz, 0);
#endif
}
#endif

// ---------------------------------------------------------------------------
// Fused attention: one workgroup per (b, block, head). 512 thr = 16 waves,
// each wave owns a 16-row strip. K/V staged to LDS by the TDM, scores in
// registers, Shaw bias via per-tile WMMA QR segment + LDS gather.
// Dynamic LDS: 64KB K + 64KB V^T + 32KB scratch = 160KB.
// ---------------------------------------------------------------------------
__global__ __launch_bounds__(512) void attn_kernel(const __bf16* __restrict__ qws,
                                                   const __bf16* __restrict__ kws,
                                                   const __bf16* __restrict__ vws,
                                                   const __bf16* __restrict__ relb,
                                                   const int* __restrict__ dists,
                                                   __bf16* __restrict__ ows) {
  extern __shared__ char smem[];
  __bf16* Kl  = (__bf16*)smem;              // [256][128]
  __bf16* Vl  = (__bf16*)(smem + 65536);    // [128][256]  (V^T)
  float*  scr = (float*)(smem + 131072);    // [16 waves][512]

  int g = blockIdx.x;
  int tid = threadIdx.x, lane = tid & 31, wv = tid >> 5;

#if HAVE_TDM
  if (wv == 0) {   // one wave drives the Tensor Data Mover for both tiles
    tdm_load_64kb(0u,      (const void*)(kws + (size_t)g * 32768));
    tdm_load_64kb(65536u,  (const void*)(vws + (size_t)g * 32768));
    __builtin_amdgcn_s_wait_tensorcnt(0);
  }
#else
  {
    const U128* ks = (const U128*)(kws + (size_t)g * 32768);
    const U128* vs = (const U128*)(vws + (size_t)g * 32768);
    U128* kd = (U128*)Kl;
    U128* vd = (U128*)Vl;
    for (int i = tid; i < 4096; i += 512) { kd[i] = ks[i]; vd[i] = vs[i]; }
  }
#endif
  __syncthreads();

  int i0 = wv * 16;
  int am = lane & 15, hl = lane >> 4;

  // Q strip fragments (16 rows x 128 K)
  const __bf16* qrow = qws + (size_t)g * 32768 + (i0 + am) * 128;
  v16bf qf[4];
#pragma unroll
  for (int kk = 0; kk < 4; kk++) qf[kk] = load_frag(qrow, kk * 32, lane);

  float* myscr = scr + wv * 512;
  float sf[16][8];  // full score strip [jtile][slot]

#pragma unroll
  for (int jt = 0; jt < 16; jt++) {
    int j0 = jt * 16;
    // Q K^T tile
    v8f qk = (v8f){0.f,0.f,0.f,0.f,0.f,0.f,0.f,0.f};
#pragma unroll
    for (int kk = 0; kk < 4; kk++) {
      v16bf bf = load_frag(Kl + (j0 + am) * 128, kk * 32, lane);
      qk = wmma_bf16(qf[kk], bf, qk);
    }
    // Shaw bias: QR segment over rel rows [rbase, rbase+32)
    int cbase = j0 - i0 + 240;
    int rbase = 257 + cbase;   // dist index of segment column 0
#pragma unroll
    for (int nt = 0; nt < 2; nt++) {
      v8f qr = (v8f){0.f,0.f,0.f,0.f,0.f,0.f,0.f,0.f};
#pragma unroll
      for (int kk = 0; kk < 4; kk++) {
        v16bf bf = load_frag(relb + (size_t)(rbase + nt * 16 + am) * 128, kk * 32, lane);
        qr = wmma_bf16(qf[kk], bf, qr);
      }
#pragma unroll
      for (int r = 0; r < 8; r++)
        myscr[(r + 8 * hl) * 32 + nt * 16 + am] = qr[r];
    }
    __builtin_amdgcn_wave_barrier();
    // gather bias via actual attention_dists, combine, scale
#pragma unroll
    for (int r = 0; r < 8; r++) {
      int i = i0 + r + 8 * hl;
      int j = j0 + am;
      int c = dists[i * 256 + j] - 257 - cbase;
      c = c < 0 ? 0 : (c > 31 ? 31 : c);
      float bias = myscr[(r + 8 * hl) * 32 + c];
      sf[jt][r] = (qk[r] + bias) * SCALE_F;
    }
    __builtin_amdgcn_wave_barrier();
  }

  // Softmax per row (rows split across 16-lane halves per C-frag layout)
#pragma unroll
  for (int r = 0; r < 8; r++) {
    float m = sf[0][r];
#pragma unroll
    for (int jt = 1; jt < 16; jt++) m = fmaxf(m, sf[jt][r]);
    m = fmaxf(m, __shfl_xor(m, 1, 32));
    m = fmaxf(m, __shfl_xor(m, 2, 32));
    m = fmaxf(m, __shfl_xor(m, 4, 32));
    m = fmaxf(m, __shfl_xor(m, 8, 32));
    float s = 0.f;
#pragma unroll
    for (int jt = 0; jt < 16; jt++) {
      float e = __expf(sf[jt][r] - m);
      sf[jt][r] = e;
      s += e;
    }
    s += __shfl_xor(s, 1, 32);
    s += __shfl_xor(s, 2, 32);
    s += __shfl_xor(s, 4, 32);
    s += __shfl_xor(s, 8, 32);
    float inv = 1.0f / s;
#pragma unroll
    for (int jt = 0; jt < 16; jt++) sf[jt][r] *= inv;
  }

  // O = P @ V  (bounce P through LDS into A-fragment layout)
  v8f oacc[8];
#pragma unroll
  for (int nt = 0; nt < 8; nt++) oacc[nt] = (v8f){0.f,0.f,0.f,0.f,0.f,0.f,0.f,0.f};

#pragma unroll
  for (int jc = 0; jc < 8; jc++) {
#pragma unroll
    for (int t = 0; t < 2; t++) {
      int jt = jc * 2 + t;
#pragma unroll
      for (int r = 0; r < 8; r++)
        myscr[(r + 8 * hl) * 32 + t * 16 + am] = sf[jt][r];
    }
    __builtin_amdgcn_wave_barrier();
    v16bf pf;
#pragma unroll
    for (int e = 0; e < 16; e++) {
      int K = 16 * (e >> 3) + 8 * hl + (e & 7);
      pf[e] = (__bf16)myscr[am * 32 + K];
    }
    __builtin_amdgcn_wave_barrier();
#pragma unroll
    for (int nt = 0; nt < 8; nt++) {
      v16bf vf = load_frag(Vl + (nt * 16 + am) * 256, jc * 32, lane);
      oacc[nt] = wmma_bf16(pf, vf, oacc[nt]);
    }
  }

  // Write O strip to [token][H*DH] bf16
  int hh = g & 7, blk = (g >> 3) & 15, b = g >> 7;
#pragma unroll
  for (int nt = 0; nt < 8; nt++) {
#pragma unroll
    for (int r = 0; r < 8; r++) {
      int i = i0 + r + 8 * hl;
      int d = nt * 16 + am;
      int tok = b * 4096 + blk * 256 + i;
      ows[(size_t)tok * 1024 + hh * 128 + d] = (__bf16)oacc[nt][r];
    }
  }
}

// ---------------------------------------------------------------------------
// Output projection: O[32768,1024] @ Wout + bout -> f32 (double-buffered)
// ---------------------------------------------------------------------------
__global__ __launch_bounds__(256) void gemm_out(const __bf16* __restrict__ ow,
                                                const __bf16* __restrict__ w,
                                                const float* __restrict__ bout,
                                                float* __restrict__ out) {
  int tid = threadIdx.x, lane = tid & 31, wv = tid >> 5;
  int m0 = blockIdx.x * 128 + wv * 16;
  int n0 = blockIdx.y * 128;
  int am = lane & 15, hl = lane >> 4;

  v8f acc[8];
#pragma unroll
  for (int nt = 0; nt < 8; nt++) acc[nt] = (v8f){0.f,0.f,0.f,0.f,0.f,0.f,0.f,0.f};

  const __bf16* arow = ow + (size_t)(m0 + am) * 1024;
  const __bf16* brow0 = w + (size_t)(n0 + am) * 1024;

  v16bf a_cur = load_frag(arow, 0, lane);
  v16bf b_cur[8];
#pragma unroll
  for (int nt = 0; nt < 8; nt++) b_cur[nt] = load_frag(brow0 + nt * 16 * 1024, 0, lane);

  for (int k0 = 0; k0 < 1024; k0 += 32) {
    v16bf a_nxt = load_frag(arow, k0 + 32, lane);
    v16bf b_nxt[8];
#pragma unroll
    for (int nt = 0; nt < 8; nt++)
      b_nxt[nt] = load_frag(brow0 + nt * 16 * 1024, k0 + 32, lane);
#pragma unroll
    for (int nt = 0; nt < 8; nt++) acc[nt] = wmma_bf16(a_cur, b_cur[nt], acc[nt]);
    a_cur = a_nxt;
#pragma unroll
    for (int nt = 0; nt < 8; nt++) b_cur[nt] = b_nxt[nt];
  }
#pragma unroll
  for (int nt = 0; nt < 8; nt++) {
    int n = n0 + nt * 16 + am;
    float bn = bout[n];
#pragma unroll
    for (int r = 0; r < 8; r++) {
      int m = m0 + r + 8 * hl;
      out[(size_t)m * 1024 + n] = acc[nt][r] + bn;
    }
  }
}

// ---------------------------------------------------------------------------
// Launch
// ---------------------------------------------------------------------------
extern "C" void kernel_launch(void* const* d_in, const int* in_sizes, int n_in,
                              void* d_out, int out_size, void* d_ws, size_t ws_size,
                              hipStream_t stream) {
  const float* hs    = (const float*)d_in[0];
  const int*   dists = (const int*)d_in[1];
  const float* gamma = (const float*)d_in[2];
  const float* beta  = (const float*)d_in[3];
  const float* Wq    = (const float*)d_in[4];
  const float* Wkv   = (const float*)d_in[5];
  const float* Wout  = (const float*)d_in[6];
  const float* bout  = (const float*)d_in[7];
  const float* rel   = (const float*)d_in[8];
  float* out = (float*)d_out;

  char* ws = (char*)d_ws;
  const size_t SZ = 67108864ULL;  // 32768*1024*2 bytes (bf16)
  __bf16* xn_bf  = (__bf16*)(ws + 0 * SZ);
  __bf16* q_ws   = (__bf16*)(ws + 1 * SZ);
  __bf16* k_ws   = (__bf16*)(ws + 2 * SZ);
  __bf16* v_ws   = (__bf16*)(ws + 3 * SZ);
  __bf16* o_ws   = (__bf16*)(ws + 4 * SZ);
  __bf16* wqkv_t = (__bf16*)(ws + 5 * SZ);
  __bf16* wout_t = (__bf16*)(ws + 5 * SZ + 6291456ULL);
  __bf16* rel_bf = (__bf16*)(ws + 5 * SZ + 6291456ULL + 2097152ULL);

  cvt_wqkv<<<(3072 * 1024 + 255) / 256, 256, 0, stream>>>(Wq, Wkv, wqkv_t);
  cvt_wout<<<(1024 * 1024 + 255) / 256, 256, 0, stream>>>(Wout, wout_t);
  cvt_rel<<<(1025 * 128 + 255) / 256, 256, 0, stream>>>(rel, rel_bf, 1025 * 128);

  ln_kernel<<<N_TOK, 256, 0, stream>>>(hs, gamma, beta, xn_bf);

  gemm_qkv<<<dim3(256, 24), 256, 0, stream>>>(xn_bf, wqkv_t, q_ws, k_ws, v_ws);

  hipFuncSetAttribute((const void*)attn_kernel,
                      hipFuncAttributeMaxDynamicSharedMemorySize, 163840);
  attn_kernel<<<NGRP, 512, 163840, stream>>>(q_ws, k_ws, v_ws, rel_bf, dists, o_ws);

  gemm_out<<<dim3(256, 8), 256, 0, stream>>>(o_ws, wout_t, bout, out);
}
